// Decoder_11871289606558
// MI455X (gfx1250) — matrix operand order repaired and checked by reading, not compiled, not run on hardware
//
#include <hip/hip_runtime.h>
#include <math.h>

#define V_  4000
#define E_  512
#define H_  512
#define N_  32
#define To_ 128
#define Ti_ 512
#define T_  129
#define SOS_ID 1
#define EOS_ID 2

typedef __attribute__((ext_vector_type(16))) __bf16 v16bf;
typedef __attribute__((ext_vector_type(8)))  __bf16 v8bf;
typedef __attribute__((ext_vector_type(8)))  float  v8f;

union V16U { v16bf v; v8bf h[2]; };

__device__ __forceinline__ __bf16 f2bf(float f) {
    union { float f; unsigned u; } v; v.f = f;
    unsigned r = (v.u + 0x7FFFu + ((v.u >> 16) & 1u)) >> 16;
    union { unsigned short s; __bf16 b; } o; o.s = (unsigned short)r;
    return o.b;
}
__device__ __forceinline__ float sigf(float x) { return 1.0f / (1.0f + __expf(-x)); }

__device__ __forceinline__ v8f wmma_bf16(v16bf a, v16bf b, v8f c) {
    return __builtin_amdgcn_wmma_f32_16x16x32_bf16(false, a, false, b, (short)0, c, false, false);
}

// A: row-major M x K bf16. 16x32 tile at (mbase, k0).
// Lane l: m = mbase + (l&15); half = l>>4.
// VGPR0-3 hold K = k0+half*8 .. +7 ; VGPR4-7 hold K = k0+16+half*8 .. +7.
__device__ __forceinline__ v16bf load_a_tile(const __bf16* __restrict__ A, int lda,
                                             int mbase, int k0, int lane) {
    int m = mbase + (lane & 15);
    int half = lane >> 4;
    const __bf16* p = A + (size_t)m * lda + k0 + half * 8;
    V16U u;
    u.h[0] = *(const v8bf*)(p);
    u.h[1] = *(const v8bf*)(p + 16);
    return u.v;
}

// B = W^T where W is (outF x K) row-major bf16. 32x16 tile: columns obase..+15, K k0..+31.
// Lane l: n = l&15 (output col), khalf = l>>4; VGPR0-3 hold K = k0+khalf*8 .. +7 ; VGPR4-7 hold K = k0+16+khalf*8 .. +7.
__device__ __forceinline__ v16bf load_b_tile(const __bf16* __restrict__ W, int ldw,
                                             int obase, int k0, int lane) {
    int n = lane & 15;
    int khalf = lane >> 4;
    const __bf16* p = W + (size_t)(obase + n) * ldw + k0 + khalf * 8;
    V16U u;
    u.h[0] = *(const v8bf*)(p);
    u.h[1] = *(const v8bf*)(p + 16);
    return u.v;
}

// ---------------- conversion / init ----------------
__global__ void cvt_kernel(const float* __restrict__ src, __bf16* __restrict__ dst, int n) {
    for (int i = blockIdx.x * blockDim.x + threadIdx.x; i < n; i += gridDim.x * blockDim.x)
        dst[i] = f2bf(src[i]);
}

__global__ void init_kernel(float* __restrict__ states, int nf, float* __restrict__ loss) {
    int i = blockIdx.x * blockDim.x + threadIdx.x;
    if (i < nf) states[i] = 0.0f;
    if (i == 0) *loss = 0.0f;
}

// ---------------- per-step prepare: embedding gather + bf16 snapshots ----------------
__global__ void prep_kernel(const int* __restrict__ tokens, int t,
                            const __bf16* __restrict__ emb_bf,
                            const float* __restrict__ att_c,
                            const float* __restrict__ h0,
                            const float* __restrict__ h1,
                            __bf16* __restrict__ rnn_in,
                            __bf16* __restrict__ h0_bf,
                            __bf16* __restrict__ h1_bf) {
    int idx = blockIdx.x * blockDim.x + threadIdx.x;     // 65536 threads
    if (idx < N_ * (E_ + H_)) {
        int n = idx >> 10, k = idx & 1023;
        if (k < E_) {
            int tok = (t == 0) ? SOS_ID : tokens[n * To_ + (t - 1)];
            rnn_in[idx] = emb_bf[(size_t)tok * E_ + k];
        } else {
            rnn_in[idx] = f2bf(att_c[n * H_ + (k - E_)]);
        }
    } else {
        int j = idx - N_ * (E_ + H_);
        if (j < N_ * H_) h0_bf[j] = f2bf(h0[j]);
        else { j -= N_ * H_; if (j < N_ * H_) h1_bf[j] = f2bf(h1[j]); }
    }
}

// ---------------- fused LSTM cell: WMMA gates + nonlinearity ----------------
// grid: 32 blocks (H column tiles), block: 128 threads (4 waves = 4 gates i,f,g,o)
__global__ void lstm_kernel(const __bf16* __restrict__ X, int Kx,
                            const __bf16* __restrict__ Hprev,
                            const __bf16* __restrict__ Wih,
                            const __bf16* __restrict__ Whh,
                            const float* __restrict__ bih,
                            const float* __restrict__ bhh,
                            float* __restrict__ c_state,
                            float* __restrict__ h_f32,
                            __bf16* __restrict__ h_bf) {
    __shared__ float gbuf[4][32][16];
    int tid  = threadIdx.x;
    int lane = tid & 31;
    int g    = tid >> 5;                 // gate index: 0=i 1=f 2=g 3=o
    int ct   = blockIdx.x;               // column tile of H (H/16 = 32)
    int o_base = g * H_ + (ct << 4);     // row into 4H-wide gate weight space

    v8f acc0 = {}, acc1 = {};
    for (int k0 = 0; k0 < Kx; k0 += 32) {
        v16bf b  = load_b_tile(Wih, Kx, o_base, k0, lane);
        v16bf a0 = load_a_tile(X, Kx, 0,  k0, lane);
        v16bf a1 = load_a_tile(X, Kx, 16, k0, lane);
        acc0 = wmma_bf16(a0, b, acc0);
        acc1 = wmma_bf16(a1, b, acc1);
    }
    for (int k0 = 0; k0 < H_; k0 += 32) {
        v16bf b  = load_b_tile(Whh, H_, o_base, k0, lane);
        v16bf a0 = load_a_tile(Hprev, H_, 0,  k0, lane);
        v16bf a1 = load_a_tile(Hprev, H_, 16, k0, lane);
        acc0 = wmma_bf16(a0, b, acc0);
        acc1 = wmma_bf16(a1, b, acc1);
    }
    int ncol = lane & 15, mh = lane >> 4;
    float bv = bih[o_base + ncol] + bhh[o_base + ncol];
#pragma unroll
    for (int r = 0; r < 8; ++r) {
        gbuf[g][mh * 8 + r][ncol]      = acc0[r] + bv;
        gbuf[g][16 + mh * 8 + r][ncol] = acc1[r] + bv;
    }
    __syncthreads();
    for (int e = tid; e < 512; e += 128) {
        int row = e >> 4, col = e & 15;      // row = batch n, col within tile
        float gi = gbuf[0][row][col], gf = gbuf[1][row][col];
        float gc = gbuf[2][row][col], go = gbuf[3][row][col];
        int idx = row * H_ + (ct << 4) + col;
        float cn = sigf(gf) * c_state[idx] + sigf(gi) * tanhf(gc);
        float hn = sigf(go) * tanhf(cn);
        c_state[idx] = cn;
        h_f32[idx] = hn;
        h_bf[idx]  = f2bf(hn);
    }
}

// ---------------- generic WMMA GEMM: out = act(X @ W^T + b) ----------------
__global__ void gemm_kernel(const __bf16* __restrict__ X, int K,
                            const __bf16* __restrict__ W,
                            const float* __restrict__ bias,
                            int Ncols, int do_tanh,
                            float* __restrict__ outF, __bf16* __restrict__ outB) {
    int lane = threadIdx.x & 31;
    int tile = blockIdx.x * (blockDim.x >> 5) + (threadIdx.x >> 5);
    int ntiles = Ncols >> 4;
    if (tile >= ntiles) return;
    int o_base = tile << 4;

    v8f acc0 = {}, acc1 = {};
    for (int k0 = 0; k0 < K; k0 += 32) {
        v16bf b  = load_b_tile(W, K, o_base, k0, lane);
        v16bf a0 = load_a_tile(X, K, 0,  k0, lane);
        v16bf a1 = load_a_tile(X, K, 16, k0, lane);
        acc0 = wmma_bf16(a0, b, acc0);
        acc1 = wmma_bf16(a1, b, acc1);
    }
    int ncol = lane & 15, mh = lane >> 4;
    float bv = bias[o_base + ncol];
#pragma unroll
    for (int r = 0; r < 8; ++r) {
        float v0 = acc0[r] + bv;
        float v1 = acc1[r] + bv;
        if (do_tanh) { v0 = tanhf(v0); v1 = tanhf(v1); }
        int row0 = mh * 8 + r, row1 = 16 + mh * 8 + r;
        size_t i0 = (size_t)row0 * Ncols + o_base + ncol;
        size_t i1 = (size_t)row1 * Ncols + o_base + ncol;
        if (outF) { outF[i0] = v0; outF[i1] = v1; }
        if (outB) { outB[i0] = f2bf(v0); outB[i1] = f2bf(v1); }
    }
}

// ---------------- attention: scores, softmax, context, mlp_in ----------------
// grid: 32 blocks (batch), block: 512 threads
__global__ void attn_kernel(const float* __restrict__ h1,
                            const float* __restrict__ enc,
                            float* __restrict__ att_c,
                            __bf16* __restrict__ mlp_in) {
    __shared__ float h1s[512];
    __shared__ float wts[512];
    __shared__ float red[512];
    int n = blockIdx.x, tid = threadIdx.x;
    h1s[tid] = h1[n * H_ + tid];
    __syncthreads();
    const float* erow = enc + ((size_t)n * Ti_ + tid) * H_;
    float sc = 0.0f;
#pragma unroll 4
    for (int k = 0; k < H_; ++k) sc += h1s[k] * erow[k];
    red[tid] = sc; __syncthreads();
    for (int s = 256; s > 0; s >>= 1) { if (tid < s) red[tid] = fmaxf(red[tid], red[tid + s]); __syncthreads(); }
    float m = red[0]; __syncthreads();
    float e = __expf(sc - m);
    red[tid] = e; __syncthreads();
    for (int s = 256; s > 0; s >>= 1) { if (tid < s) red[tid] += red[tid + s]; __syncthreads(); }
    float inv = 1.0f / red[0]; __syncthreads();
    wts[tid] = e * inv;
    __syncthreads();
    float acc = 0.0f;
    const float* ebase = enc + (size_t)n * Ti_ * H_ + tid;
#pragma unroll 4
    for (int tt = 0; tt < Ti_; ++tt) acc += wts[tt] * ebase[(size_t)tt * H_];
    att_c[n * H_ + tid] = acc;
    mlp_in[n * (2 * H_) + tid]       = f2bf(h1s[tid]);
    mlp_in[n * (2 * H_) + H_ + tid]  = f2bf(acc);
}

// ---------------- NLL over V per row, accumulate ----------------
__global__ void loss_kernel(const float* __restrict__ logits,
                            const int* __restrict__ tokens, int t,
                            float* __restrict__ loss_acc) {
    __shared__ float red[256];
    int n = blockIdx.x, tid = threadIdx.x;
    const float* lg = logits + (size_t)n * V_;
    float m = -3.4e38f;
    for (int j = tid; j < V_; j += 256) m = fmaxf(m, lg[j]);
    red[tid] = m; __syncthreads();
    for (int s = 128; s > 0; s >>= 1) { if (tid < s) red[tid] = fmaxf(red[tid], red[tid + s]); __syncthreads(); }
    m = red[0]; __syncthreads();
    float sm = 0.0f;
    for (int j = tid; j < V_; j += 256) sm += __expf(lg[j] - m);
    red[tid] = sm; __syncthreads();
    for (int s = 128; s > 0; s >>= 1) { if (tid < s) red[tid] += red[tid + s]; __syncthreads(); }
    if (tid == 0) {
        int target = (t < To_) ? tokens[n * To_ + t] : EOS_ID;
        float nll = logf(red[0]) + m - lg[target];
        atomicAdd(loss_acc, nll);
    }
}

__global__ void final_kernel(const float* __restrict__ loss_acc, float* __restrict__ out) {
    if (threadIdx.x == 0 && blockIdx.x == 0)
        out[0] = loss_acc[0] * ((float)(T_ - 1) / (float)(N_ * T_));
}

// ---------------------------------------------------------------------------
extern "C" void kernel_launch(void* const* d_in, const int* in_sizes, int n_in,
                              void* d_out, int out_size, void* d_ws, size_t ws_size,
                              hipStream_t stream) {
    const int*   padded = (const int*)  d_in[0];
    const float* enc    = (const float*)d_in[1];
    const float* emb    = (const float*)d_in[2];
    const float* W_ih0  = (const float*)d_in[3];
    const float* b_ih0  = (const float*)d_in[4];
    const float* W_hh0  = (const float*)d_in[5];
    const float* b_hh0  = (const float*)d_in[6];
    const float* W_ih1  = (const float*)d_in[7];
    const float* b_ih1  = (const float*)d_in[8];
    const float* W_hh1  = (const float*)d_in[9];
    const float* b_hh1  = (const float*)d_in[10];
    const float* W1     = (const float*)d_in[11];
    const float* b1     = (const float*)d_in[12];
    const float* W2     = (const float*)d_in[13];
    const float* b2     = (const float*)d_in[14];

    char* base = (char*)d_ws;
    size_t off = 0;
    auto alloc = [&](size_t bytes) -> void* {
        void* p = base + off;
        off += (bytes + 255) & ~(size_t)255;
        return p;
    };

    __bf16* wih0_bf = (__bf16*)alloc((size_t)4 * H_ * (E_ + H_) * 2);  // 2048x1024
    __bf16* whh0_bf = (__bf16*)alloc((size_t)4 * H_ * H_ * 2);         // 2048x512
    __bf16* wih1_bf = (__bf16*)alloc((size_t)4 * H_ * H_ * 2);
    __bf16* whh1_bf = (__bf16*)alloc((size_t)4 * H_ * H_ * 2);
    __bf16* w1_bf   = (__bf16*)alloc((size_t)H_ * 2 * H_ * 2);         // 512x1024
    __bf16* w2_bf   = (__bf16*)alloc((size_t)V_ * H_ * 2);             // 4000x512
    __bf16* emb_bf  = (__bf16*)alloc((size_t)V_ * E_ * 2);             // 4000x512
    float*  states  = (float*) alloc((size_t)5 * N_ * H_ * 4);
    float* h0 = states, *c0 = states + N_ * H_, *h1 = states + 2 * N_ * H_;
    float* c1 = states + 3 * N_ * H_, *attc = states + 4 * N_ * H_;
    __bf16* rnn_in = (__bf16*)alloc((size_t)N_ * (E_ + H_) * 2);
    __bf16* h0p_bf = (__bf16*)alloc((size_t)N_ * H_ * 2);
    __bf16* h1p_bf = (__bf16*)alloc((size_t)N_ * H_ * 2);
    __bf16* h0n_bf = (__bf16*)alloc((size_t)N_ * H_ * 2);
    __bf16* h1n_bf = (__bf16*)alloc((size_t)N_ * H_ * 2);
    __bf16* mlp_in = (__bf16*)alloc((size_t)N_ * 2 * H_ * 2);
    __bf16* hidden = (__bf16*)alloc((size_t)N_ * H_ * 2);
    float*  logits = (float*) alloc((size_t)N_ * V_ * 4);
    float*  loss   = (float*) alloc(256);

    auto cvt = [&](const float* s, __bf16* d, int n) {
        int blocks = (n + 1023) / 1024;
        if (blocks > 2048) blocks = 2048;
        cvt_kernel<<<blocks, 256, 0, stream>>>(s, d, n);
    };
    cvt(W_ih0, wih0_bf, 4 * H_ * (E_ + H_));
    cvt(W_hh0, whh0_bf, 4 * H_ * H_);
    cvt(W_ih1, wih1_bf, 4 * H_ * H_);
    cvt(W_hh1, whh1_bf, 4 * H_ * H_);
    cvt(W1,    w1_bf,   H_ * 2 * H_);
    cvt(W2,    w2_bf,   V_ * H_);
    cvt(emb,   emb_bf,  V_ * E_);

    int nstates = 5 * N_ * H_;
    init_kernel<<<(nstates + 255) / 256, 256, 0, stream>>>(states, nstates, loss);

    for (int t = 0; t < T_; ++t) {
        // embedding gather + att_c concat + bf16 snapshots of h0/h1
        prep_kernel<<<256, 256, 0, stream>>>(padded, t, emb_bf, attc, h0, h1,
                                             rnn_in, h0p_bf, h1p_bf);
        // LSTM layer 0: X = [emb | att_c] (K=1024), Hprev = h0
        lstm_kernel<<<H_ / 16, 128, 0, stream>>>(rnn_in, E_ + H_, h0p_bf,
                                                 wih0_bf, whh0_bf, b_ih0, b_hh0,
                                                 c0, h0, h0n_bf);
        // LSTM layer 1: X = h0_new (K=512), Hprev = h1
        lstm_kernel<<<H_ / 16, 128, 0, stream>>>(h0n_bf, H_, h1p_bf,
                                                 wih1_bf, whh1_bf, b_ih1, b_hh1,
                                                 c1, h1, h1n_bf);
        // dot-product attention + softmax + context + mlp_in = [h1 | ctx]
        attn_kernel<<<N_, 512, 0, stream>>>(h1, enc, attc, mlp_in);
        // hidden = tanh(mlp_in @ W1^T + b1): 32 tiles -> 4 blocks x 8 waves
        gemm_kernel<<<4, 256, 0, stream>>>(mlp_in, 2 * H_, w1_bf, b1, H_, 1,
                                           (float*)nullptr, hidden);
        // logits = hidden @ W2^T + b2: 250 tiles -> 32 blocks x 8 waves
        gemm_kernel<<<32, 256, 0, stream>>>(hidden, H_, w2_bf, b2, V_, 0,
                                            logits, (__bf16*)nullptr);
        // log-softmax NLL accumulate
        loss_kernel<<<N_, 256, 0, stream>>>(logits, padded, t, loss);
    }
    final_kernel<<<1, 32, 0, stream>>>(loss, (float*)d_out);
}
